// RemoveDuplicateFeaturesEncoderStep_14181982012124
// MI455X (gfx1250) — matrix operand order repaired and checked by reading, not compiled, not run on hardware
//
#include <hip/hip_runtime.h>
#include <hip/hip_bf16.h>
#include <stddef.h>

// Problem constants (from reference)
#define SEQ   2048
#define BATCH 128
#define FEAT  256
#define SEP   1024            // single_eval_pos
#define BF    (BATCH * FEAT)  // 32768
#define NSUP  10              // 4x4 feature super-tiles (64x64), upper triangle
#define KC    128             // K-chunk staged in LDS per iteration
#define LDST  136             // LDS row stride in halfs (128 data + 8 pad -> 68-dword stride, conflict-free)

typedef __attribute__((ext_vector_type(8)))  _Float16 v8h;
typedef __attribute__((ext_vector_type(16))) _Float16 v16h;
typedef __attribute__((ext_vector_type(8)))  float    v8f;

// ---------------------------------------------------------------------------
// Async global->LDS copy, 16 bytes per lane (ASYNCcnt-tracked CDNA5 path).
// ---------------------------------------------------------------------------
__device__ __forceinline__ void async_copy_b128(unsigned lds_off, const void* gaddr) {
    asm volatile("global_load_async_to_lds_b128 %0, %1, off"
                 :: "v"(lds_off), "v"(gaddr) : "memory");
}
__device__ __forceinline__ void wait_asynccnt0() {
    asm volatile("s_wait_asynccnt 0x0" ::: "memory");
}
__device__ __forceinline__ unsigned lds_off_of(const void* p) {
    return (unsigned)(unsigned long long)p;   // low 32 bits of LDS aperture addr = byte offset
}

// ---------------------------------------------------------------------------
// Kernel 1: per-(b,f) stats over s=0..SEP-1.
//   mean, inv = 1/(std_ddof1 + 1e-6), mean2 = mean of clip((x-mean)*inv)
//   Also zero-init colsum. One thread per (b,f); coalesced column reduction.
// ---------------------------------------------------------------------------
__global__ void k_stats(const float* __restrict__ x,
                        float* __restrict__ mean_o,
                        float* __restrict__ inv_o,
                        float* __restrict__ mean2_o,
                        int* __restrict__ colsum) {
    int t = blockIdx.x * blockDim.x + threadIdx.x;   // 0..BF-1
    if (t >= BF) return;
    colsum[t] = 0;

    float s1 = 0.f, s2 = 0.f;
    for (int s = 0; s < SEP; ++s) {
        float v = x[(size_t)s * BF + t];
        s1 += v;
        s2 += v * v;
    }
    float m   = s1 * (1.0f / SEP);
    float var = (s2 - (float)SEP * m * m) * (1.0f / (SEP - 1));
    var = fmaxf(var, 0.0f);
    float inv = 1.0f / (sqrtf(var) + 1e-6f);

    float s3 = 0.f;
    for (int s = 0; s < SEP; ++s) {
        float v  = x[(size_t)s * BF + t];
        float xn = fminf(fmaxf((v - m) * inv, -100.0f), 100.0f);
        s3 += xn;
    }
    mean_o[t]  = m;
    inv_o[t]   = inv;
    mean2_o[t] = s3 * (1.0f / SEP);
}

// ---------------------------------------------------------------------------
// Kernel 2: pack XcT[b][f][s] = f16( clip((x-mean)*inv) - mean2 ).
// One thread per (b,f); coalesced reads, register-buffered 32B contiguous
// writes. XcT (64 MB) stays L2-resident for the cov kernel.
// ---------------------------------------------------------------------------
__global__ void k_pack(const float* __restrict__ x,
                       const float* __restrict__ mean_i,
                       const float* __restrict__ inv_i,
                       const float* __restrict__ mean2_i,
                       _Float16* __restrict__ xct) {
    int t = blockIdx.x * blockDim.x + threadIdx.x;   // 0..BF-1
    if (t >= BF) return;
    float m  = mean_i[t];
    float iv = inv_i[t];
    float m2 = mean2_i[t];
    _Float16* dst = xct + (size_t)t * SEP;           // t == b*FEAT + f

    for (int sc = 0; sc < SEP; sc += 32) {
        v16h lo, hi;
#pragma unroll
        for (int j = 0; j < 16; ++j) {
            float a  = x[(size_t)(sc + j) * BF + t];
            float bb = x[(size_t)(sc + 16 + j) * BF + t];
            lo[j] = (_Float16)(fminf(fmaxf((a  - m) * iv, -100.f), 100.f) - m2);
            hi[j] = (_Float16)(fminf(fmaxf((bb - m) * iv, -100.f), 100.f) - m2);
        }
        *(v16h*)(dst + sc)      = lo;
        *(v16h*)(dst + sc + 16) = hi;
    }
}

// ---------------------------------------------------------------------------
// Kernel 3: LDS-blocked cov via WMMA f16 -> f32, threshold, triu col-sum.
// Block (8 waves) owns one (batch, 64x64 upper-tri super-tile). Per K-chunk
// of 128, A/B feature slabs (64x128 halfs each) are staged into LDS with
// global_load_async_to_lds_b128 (s_wait_asynccnt + barrier), then each wave
// computes a 16x32 output region: one shared A tile x two B tiles,
// 2 accumulators, 8 x v_wmma_f32_16x16x32_f16 per chunk.
//   A lane layout (16-bit 16x32): lane<16 K{0..15}, lane>=16 K{16..31} (contiguous, same as B)
//   B lane layout (32x16): lane holds col N=lane%16, 16 consecutive K (+16 if lane>=16)
// cov > 0.999  <=>  dot > 0.999*(SEP-1).
// ---------------------------------------------------------------------------
__global__ void k_cov(const _Float16* __restrict__ xct,
                      int* __restrict__ colsum) {
    __shared__ alignas(32) _Float16 sA[64][LDST];
    __shared__ alignas(32) _Float16 sB[64][LDST];

    const int b    = blockIdx.x / NSUP;
    int st = blockIdx.x % NSUP;
    // decode 4x4 upper-tri super-tile index -> (si, sj), si <= sj
    int si = 0;
    while (st >= (4 - si)) { st -= (4 - si); ++si; }
    const int sj = si + st;

    const int wave = threadIdx.x >> 5;
    const int lane = threadIdx.x & 31;
    const int r    = lane & 15;
    const int hs   = lane >> 4;             // 0: lanes 0-15, 1: lanes 16-31
    const int ai   = wave & 3;              // A tile row within super-tile
    const int bj0  = (wave >> 2) * 2;       // first of two B tiles

    const _Float16* gbase = xct + (size_t)b * FEAT * SEP;
    const int tr = threadIdx.x >> 4;        // staging row lane group 0..15
    const int tc = threadIdx.x & 15;        // staging 16B column 0..15

    v8f c0 = {}, c1 = {};
    for (int k0 = 0; k0 < SEP; k0 += KC) {
        __syncthreads();                    // LDS safe to overwrite
        // stage 128 feature-rows x 128 halfs (A slab then B slab), 8 passes
#pragma unroll
        for (int p = 0; p < 8; ++p) {
            int row  = p * 16 + tr;         // 0..127
            int isA  = (row < 64);
            int frow = isA ? row : (row - 64);
            int feat = (isA ? si : sj) * 64 + frow;
            const _Float16* g = gbase + (size_t)feat * SEP + k0 + tc * 8;
            unsigned l = isA ? lds_off_of(&sA[frow][tc * 8])
                             : lds_off_of(&sB[frow][tc * 8]);
            async_copy_b128(l, g);
        }
        wait_asynccnt0();
        __syncthreads();                    // slabs visible to all waves

#pragma unroll
        for (int kk = 0; kk < KC; kk += 32) {
            v8h a0 = *(const v8h*)&sA[ai * 16 + r][kk + hs * 16];
            v8h a1 = *(const v8h*)&sA[ai * 16 + r][kk + hs * 16 + 8];
            v16h a = __builtin_shufflevector(a0, a1,
                       0, 1, 2, 3, 4, 5, 6, 7, 8, 9, 10, 11, 12, 13, 14, 15);
            v8h b0lo = *(const v8h*)&sB[bj0 * 16 + r][kk + hs * 16];
            v8h b0hi = *(const v8h*)&sB[bj0 * 16 + r][kk + hs * 16 + 8];
            v16h bm0 = __builtin_shufflevector(b0lo, b0hi,
                       0, 1, 2, 3, 4, 5, 6, 7, 8, 9, 10, 11, 12, 13, 14, 15);
            v8h b1lo = *(const v8h*)&sB[(bj0 + 1) * 16 + r][kk + hs * 16];
            v8h b1hi = *(const v8h*)&sB[(bj0 + 1) * 16 + r][kk + hs * 16 + 8];
            v16h bm1 = __builtin_shufflevector(b1lo, b1hi,
                       0, 1, 2, 3, 4, 5, 6, 7, 8, 9, 10, 11, 12, 13, 14, 15);
            c0 = __builtin_amdgcn_wmma_f32_16x16x32_f16(
                     false, a, false, bm0, (short)0, c0, false, false);
            c1 = __builtin_amdgcn_wmma_f32_16x16x32_f16(
                     false, a, false, bm1, (short)0, c1, false, false);
        }
    }

    // D layout: lane l, VGPR v: row M = v + 8*(l>=16), col N = l%16
    const float thr = 0.999f * (float)(SEP - 1);
    const int gi_base = si * 64 + ai * 16 + hs * 8;
    const int gj0 = sj * 64 + bj0 * 16 + r;
    const int gj1 = gj0 + 16;
    int cnt0 = 0, cnt1 = 0;
#pragma unroll
    for (int v = 0; v < 8; ++v) {
        int gi = gi_base + v;
        if (gi <= gj0 && c0[v] > thr) ++cnt0;
        if (gi <= gj1 && c1[v] > thr) ++cnt1;
    }
    if (cnt0) atomicAdd(&colsum[b * FEAT + gj0], cnt0);
    if (cnt1) atomicAdd(&colsum[b * FEAT + gj1], cnt1);
}

// ---------------------------------------------------------------------------
// Kernel 4: stable order per batch. sel[f] = (colsum==1). Selected features
// first (original order), then the rest. One block (256 thr) per batch.
// ---------------------------------------------------------------------------
__global__ void k_order(const int* __restrict__ colsum,
                        int* __restrict__ order,
                        int* __restrict__ counts) {
    __shared__ int scan[FEAT];
    const int b = blockIdx.x;
    const int f = threadIdx.x;
    const int sel = (colsum[b * FEAT + f] == 1) ? 1 : 0;

    scan[f] = sel;
    __syncthreads();
    for (int off = 1; off < FEAT; off <<= 1) {
        int v = scan[f];
        int add = (f >= off) ? scan[f - off] : 0;
        __syncthreads();
        scan[f] = v + add;
        __syncthreads();
    }
    const int inc   = scan[f];
    const int total = scan[FEAT - 1];
    const int pos = sel ? (inc - 1) : (total + f - inc);
    order[b * FEAT + pos] = f;
    if (f == 0) counts[b] = total;
}

// ---------------------------------------------------------------------------
// Kernel 5: gather + mask. out[s,b,f] = f<count ? x[s,b,order[b,f]] : 0
// ---------------------------------------------------------------------------
__global__ void k_gather(const float* __restrict__ x,
                         const int* __restrict__ order,
                         const int* __restrict__ counts,
                         float* __restrict__ out) {
    size_t idx = (size_t)blockIdx.x * blockDim.x + threadIdx.x;
    if (idx >= (size_t)SEQ * BF) return;
    int f = (int)(idx & (FEAT - 1));
    int b = (int)((idx >> 8) & (BATCH - 1));
    size_t sb = idx >> 8;                    // s*BATCH + b
    int cnt = counts[b];
    int src = order[b * FEAT + f];
    float v = x[sb * FEAT + src];
    out[idx] = (f < cnt) ? v : 0.0f;
}

// ---------------------------------------------------------------------------
extern "C" void kernel_launch(void* const* d_in, const int* in_sizes, int n_in,
                              void* d_out, int out_size, void* d_ws, size_t ws_size,
                              hipStream_t stream) {
    const float* x = (const float*)d_in[0];
    float* out = (float*)d_out;

    // workspace layout
    char* w = (char*)d_ws;
    _Float16* xct = (_Float16*)w;                          // 128*256*1024*2 = 64 MB
    float* mean   = (float*)(w + (size_t)BATCH * FEAT * SEP * 2);
    float* inv    = mean  + BF;
    float* mean2  = inv   + BF;
    int*   colsum = (int*)(mean2 + BF);
    int*   order  = colsum + BF;
    int*   counts = order  + BF;
    (void)in_sizes; (void)n_in; (void)out_size; (void)ws_size;

    k_stats<<<BF / 256, 256, 0, stream>>>(x, mean, inv, mean2, colsum);
    k_pack <<<BF / 256, 256, 0, stream>>>(x, mean, inv, mean2, xct);
    // 128 batches * 10 upper-tri 64x64 super-tiles, 8 waves each
    k_cov  <<<BATCH * NSUP, 256, 0, stream>>>(xct, colsum);
    k_order<<<BATCH, FEAT, 0, stream>>>(colsum, order, counts);
    k_gather<<<(unsigned)(((size_t)SEQ * BF) / 256), 256, 0, stream>>>(x, order, counts, out);
}
